// PersistenceImageEncoder_16269336117486
// MI455X (gfx1250) — matrix operand order repaired and checked by reading, not compiled, not run on hardware
//
#include <hip/hip_runtime.h>

typedef _Float16 v2h  __attribute__((ext_vector_type(2)));
typedef _Float16 v8h  __attribute__((ext_vector_type(8)));
typedef _Float16 v16h __attribute__((ext_vector_type(16)));
typedef float    v8f  __attribute__((ext_vector_type(8)));

#define HH 50
#define WW 50
#define EPSF 1e-8f
#define KCHUNK 64            // points per chunk (two 16x16x32 WMMAs)
#define TPB 512              // 16 waves (wave32)
#define LOG2E 1.44269504088896340736f
#define BIGGRID 1.0e4f       // poisoned grid value for padded rows -> exp2 == 0

// ---------------------------------------------------------------------------
// image[h,w] = sum_n pers_n * gy[n,h] * gx[n,w]  ==  A^T B rank-2M GEMM.
// Pad 50 -> 64. Per 64-point chunk:
//   phase A: 512 threads fill As[64][64] (pers-weighted gy) and Bs[64][64]
//            (gx) in f16. Branch-free: wave-uniform A/B split, padded rows
//            poisoned so exp2 underflows to 0, quadratic expanded to FMA form,
//            two adjacent points per thread -> packed ds_store_b32 and one
//            float4 (global_load_b128) pair load.
//   phase B: 16 waves x one 16x16 tile, two v_wmma_f32_16x16x32_f16 each.
// Double-buffered LDS -> single barrier per chunk. K-split over blocks,
// f32 atomicAdd reduction into workspace, then max-normalize.
// ---------------------------------------------------------------------------
__global__ void __launch_bounds__(TPB) pimg_accum(
    const float4* __restrict__ pairs4,   // two (birth, death) points per elem
    const float*  __restrict__ sigma_p,
    const float*  __restrict__ grid_x,
    const float*  __restrict__ grid_y,
    float*        __restrict__ accum,
    int nPoints, int nChunks)
{
    __shared__ __align__(16) _Float16 As[2][64][KCHUNK];   // As[b][h][k]
    __shared__ __align__(16) _Float16 Bs[2][64][KCHUNK];   // Bs[b][w][k]

    const int t    = threadIdx.x;
    const int kloc = t & 31;
    const int lane = t & 31;
    const int g    = __builtin_amdgcn_readfirstlane(t >> 5);  // wave id, scalar
    const int tm   = g >> 2;                                  // tile row
    const int tn   = g & 3;                                   // tile col

    const float sigma  = sigma_p[0];
    const float inv2s2 = 1.0f / (2.0f * sigma * sigma);
    const float nrm    = 1.0f / (1.0f + EPSF);
    const float negc   = -inv2s2 * LOG2E;       // exp(x) = exp2(x*log2(e))

    // Wave-uniform: waves 0-7 produce As rows, waves 8-15 produce Bs rows.
    const bool isA  = (g < 8);
    const int rbase = (g & 7) * 8;              // 8 consecutive rows per thread

    // Per-row quadratic coefficients: negc*(gv-ctr)^2 = (qb*ctr + qa) + negc*ctr^2
    // Padded rows get gv = BIGGRID so the exponent is ~-7e9 -> exp2 -> exactly 0.
    const float* grid = isA ? grid_y : grid_x;
    float qa[8], qb[8];
#pragma unroll
    for (int j = 0; j < 8; ++j) {
        int   r  = rbase + j;
        float gv = (r < HH) ? grid[r] : BIGGRID;    // HH == WW == 50
        qa[j] = negc * gv * gv;
        qb[j] = -2.0f * negc * gv;
    }

    // This thread owns chunk slots 2*kloc and 2*kloc+1 (adjacent -> b32 store).
    _Float16* dst = (isA ? &As[0][0][0] : &Bs[0][0][0]) + rbase * KCHUNK + 2 * kloc;
    const int bufStride = 64 * KCHUNK;          // halfs per buffer

    const int nP4 = nPoints >> 1;               // full float4 pair-of-points
    v8f acc = {};
    int buf = 0;

    for (int c = blockIdx.x; c < nChunks; c += gridDim.x, buf ^= 1) {
        // ---- phase A: branch-free Gaussian fill ----
        const int i4  = c * 32 + kloc;                       // float4 index
        const int i4c = max(min(i4, nP4 - 1), 0);
        const float4 pp = pairs4[i4c];                       // global_load_b128
        const float pers0 = pp.y - pp.x;
        const float pers1 = pp.w - pp.z;
        const float m0 = (2 * i4     < nPoints) ? 1.0f : 0.0f;
        const float m1 = (2 * i4 + 1 < nPoints) ? 1.0f : 0.0f;

        // center: pers_norm for A rows, birth_norm for B rows (select, no branch)
        const float ctr0 = (isA ? pers0 : pp.x) * nrm;
        const float ctr1 = (isA ? pers1 : pp.z) * nrm;
        // post-exp scale: masked weight (pers) for A rows, mask for B rows
        const float s0 = isA ? pers0 * m0 : m0;
        const float s1 = isA ? pers1 * m1 : m1;
        const float qc0 = negc * ctr0 * ctr0;
        const float qc1 = negc * ctr1 * ctr1;

        // speculative prefetch of next chunk's pairs (clamped, branch-free)
        {
            int kn = (c + (int)gridDim.x) * 32 + kloc;
            __builtin_prefetch(&pairs4[kn < nP4 ? kn : 0], 0, 1);
        }

        _Float16* d = dst + buf * bufStride;
#pragma unroll
        for (int j = 0; j < 8; ++j) {
            float t0 = fmaf(qb[j], ctr0, qa[j]) + qc0;
            float t1 = fmaf(qb[j], ctr1, qa[j]) + qc1;
            float v0 = s0 * __builtin_amdgcn_exp2f(t0);
            float v1 = s1 * __builtin_amdgcn_exp2f(t1);
            v2h pk = { (_Float16)v0, (_Float16)v1 };
            *(v2h*)(d + j * KCHUNK) = pk;                    // ds_store_b32
        }

        __syncthreads();   // double-buffered: one barrier per chunk suffices

        // ---- phase B: two 16x16x32 WMMAs per wave ----
        // A frag: lanes 0-15: M=lane,    K = 0..7  & 16..23
        //         lanes16-31: M=lane-16, K = 8..15 & 24..31
        // B frag: lanes 0-15: N=lane,    K = 0..7  & 16..23
        //         lanes16-31: N=lane-16, K = 8..15 & 24..31
        const int m   = lane & 15;
        const int k0a = (lane < 16) ? 0 : 8;
        const int k0b = (lane < 16) ? 0 : 8;
        const _Float16* arow = &As[buf][tm * 16 + m][0];
        const _Float16* brow = &Bs[buf][tn * 16 + m][0];
#pragma unroll
        for (int kh = 0; kh < 2; ++kh) {
            v8h alo = *(const v8h*)(arow + kh * 32 + k0a);
            v8h ahi = *(const v8h*)(arow + kh * 32 + k0a + 16);
            v16h a = __builtin_shufflevector(alo, ahi,
                     0,1,2,3,4,5,6,7,8,9,10,11,12,13,14,15);
            v8h blo = *(const v8h*)(brow + kh * 32 + k0b);
            v8h bhi = *(const v8h*)(brow + kh * 32 + k0b + 16);
            v16h b = __builtin_shufflevector(blo, bhi,
                     0,1,2,3,4,5,6,7,8,9,10,11,12,13,14,15);
            acc = __builtin_amdgcn_wmma_f32_16x16x32_f16(
                      false, a, false, b, (short)0, acc, false, false);
        }
    }

    // ---- K-split reduction into global image ----
    // C/D layout: VGPR r, lanes 0-15 -> M=r, lanes 16-31 -> M=r+8; N=lane&15.
    const int n     = lane & 15;
    const int hbase = tm * 16 + ((lane >> 4) << 3);
    const int w     = tn * 16 + n;
#pragma unroll
    for (int r = 0; r < 8; ++r) {
        int h = hbase + r;
        if (h < HH && w < WW) atomicAdd(&accum[h * WW + w], acc[r]);
    }
}

// ---------------------------------------------------------------------------
__global__ void zero_accum(float* __restrict__ accum, int n)
{
    int i = blockIdx.x * blockDim.x + threadIdx.x;
    if (i < n) accum[i] = 0.0f;
}

// Single-block max-reduce + normalize: out[i] = accum[i] / (max + EPS)
__global__ void __launch_bounds__(1024) pimg_finalize(
    const float* __restrict__ accum, float* __restrict__ out, int n)
{
    __shared__ float red[1024];
    float m = -3.4e38f;
    for (int i = threadIdx.x; i < n; i += 1024) m = fmaxf(m, accum[i]);
    red[threadIdx.x] = m;
    __syncthreads();
#pragma unroll
    for (int s = 512; s > 0; s >>= 1) {
        if (threadIdx.x < (unsigned)s)
            red[threadIdx.x] = fmaxf(red[threadIdx.x], red[threadIdx.x + s]);
        __syncthreads();
    }
    float mx = red[0];
    for (int i = threadIdx.x; i < n; i += 1024) out[i] = accum[i] / (mx + EPSF);
}

// ---------------------------------------------------------------------------
extern "C" void kernel_launch(void* const* d_in, const int* in_sizes, int n_in,
                              void* d_out, int out_size, void* d_ws, size_t ws_size,
                              hipStream_t stream)
{
    const float4* pairs4 = (const float4*)d_in[0];
    const float*  sigma  = (const float*)d_in[1];
    const float*  grid_x = (const float*)d_in[2];
    const float*  grid_y = (const float*)d_in[3];
    float* out   = (float*)d_out;
    float* accum = (float*)d_ws;        // 2500 f32 accumulation image

    const int nPoints = in_sizes[0] / 2;
    const int nChunks = (nPoints + KCHUNK - 1) / KCHUNK;
    const int nPix    = HH * WW;

    int grid = nChunks < 1024 ? nChunks : 1024;
    if (grid < 1) grid = 1;

    zero_accum<<<(nPix + 255) / 256, 256, 0, stream>>>(accum, nPix);
    pimg_accum<<<grid, TPB, 0, stream>>>(pairs4, sigma, grid_x, grid_y,
                                         accum, nPoints, nChunks);
    pimg_finalize<<<1, 1024, 0, stream>>>(accum, out, nPix);
}
